// Mamba_78683800862945
// MI455X (gfx1250) — matrix operand order repaired and checked by reading, not compiled, not run on hardware
//
#include <hip/hip_runtime.h>

// ---------------------------------------------------------------------------
// Mamba block for MI455X (gfx1250): all GEMMs on v_wmma_f32_16x16x32_f16.
// LDS B-tiles stored transposed so every WMMA fragment is 2x ds_load_b128.
// Staging: fixed-trip unrolled loops, all global loads issued before the
// convert+ds_store phase (no per-element loadcnt round trip).
// Chunk-parallel selective scan. Wave32 / WGP model.
// ---------------------------------------------------------------------------

#define L_DIM   1024
#define B_DIM   2
#define F_DIM   1024
#define H_DIM   2048          // E * F
#define N_DIM   16
#define R_DIM   64
#define K_CONV  4
#define MROWS   (L_DIM * B_DIM)   // 2048 flattened (l,b) rows
#define PROJ_W  96                // R + 2N

#define NCHUNK  16
#define CLEN    (L_DIM / NCHUNK)  // 64
#define BHN     (B_DIM * H_DIM * N_DIM)  // 65536

// WMMA tiling: 256 threads = 8 wave32 (4x2), block tile 128x128,
// wave tile 32x64 = 2x4 WMMA tiles, K-chunk 32.
#define BM 128
#define BN 128
#define KC 32
#define LDA_S (KC + 8)   // A tile row-major   [BM][LDA_S]
#define LDB_T (KC + 8)   // B tile TRANSPOSED  [BN][LDB_T]  (col-major)
#define NTHREADS 256

typedef __attribute__((ext_vector_type(16))) _Float16 v16h;
typedef __attribute__((ext_vector_type(8)))  _Float16 v8h;
typedef __attribute__((ext_vector_type(4)))  _Float16 v4h;
typedef __attribute__((ext_vector_type(8)))  float    v8f;
typedef __attribute__((ext_vector_type(4)))  float    v4f;

__device__ __forceinline__ float siluf(float v) {
  return v / (1.0f + __expf(-v));
}
__device__ __forceinline__ float softplusf(float v) {
  return (v > 20.0f) ? v : log1pf(__expf(v));
}

// A fragment: 16x32 f16, lane = row (M). Lanes 0-15 hold
// K 0..15, lanes 16-31 hold K 16..31 (matches B). Two b128 loads.
__device__ __forceinline__ v16h frag_a(const _Float16* As, int rowBase) {
  const int lane = threadIdx.x & 31;
  const int hlf = lane >> 4, m = lane & 15;
  const _Float16* p = As + (size_t)(rowBase + m) * LDA_S + hlf * 16;
  v16h a;
#pragma unroll
  for (int i = 0; i < 8; ++i) {
    a[i]     = p[i];
    a[8 + i] = p[8 + i];
  }
  return a;
}

// B fragment: 32x16 f16 (KxN), lane = column (N); lanes 0-15 hold K 0..15,
// lanes 16-31 hold K 16..31. LDS tile is transposed [col][k], so each lane
// reads 16 contiguous halves (32B, 16B-aligned) -> 2x ds_load_b128.
__device__ __forceinline__ v16h frag_b_t(const _Float16* Bs_t, int colBase) {
  const int lane = threadIdx.x & 31;
  const int hlf = lane >> 4, n = lane & 15;
  const _Float16* p = Bs_t + (size_t)(colBase + n) * LDB_T + hlf * 16;
  v16h b;
#pragma unroll
  for (int i = 0; i < 16; ++i) b[i] = p[i];
  return b;
}

// One K-step: wave owns a 32x64 tile at (waveM*32, waveN*64); 2x4 WMMAs.
__device__ __forceinline__ void wmma_step(const _Float16* As, const _Float16* Bs_t,
                                          v8f acc[2][4]) {
  const int wave = threadIdx.x >> 5;
  const int wm = (wave >> 1) * 32, wn = (wave & 1) * 64;
  v16h a0 = frag_a(As, wm);
  v16h a1 = frag_a(As, wm + 16);
  v16h b0 = frag_b_t(Bs_t, wn);
  v16h b1 = frag_b_t(Bs_t, wn + 16);
  v16h b2 = frag_b_t(Bs_t, wn + 32);
  v16h b3 = frag_b_t(Bs_t, wn + 48);
  acc[0][0] = __builtin_amdgcn_wmma_f32_16x16x32_f16(false, a0, false, b0, (short)0, acc[0][0], false, false);
  acc[0][1] = __builtin_amdgcn_wmma_f32_16x16x32_f16(false, a0, false, b1, (short)0, acc[0][1], false, false);
  acc[0][2] = __builtin_amdgcn_wmma_f32_16x16x32_f16(false, a0, false, b2, (short)0, acc[0][2], false, false);
  acc[0][3] = __builtin_amdgcn_wmma_f32_16x16x32_f16(false, a0, false, b3, (short)0, acc[0][3], false, false);
  acc[1][0] = __builtin_amdgcn_wmma_f32_16x16x32_f16(false, a1, false, b0, (short)0, acc[1][0], false, false);
  acc[1][1] = __builtin_amdgcn_wmma_f32_16x16x32_f16(false, a1, false, b1, (short)0, acc[1][1], false, false);
  acc[1][2] = __builtin_amdgcn_wmma_f32_16x16x32_f16(false, a1, false, b2, (short)0, acc[1][2], false, false);
  acc[1][3] = __builtin_amdgcn_wmma_f32_16x16x32_f16(false, a1, false, b3, (short)0, acc[1][3], false, false);
}

// Stage A tile (f32 source, row-major, stride lda) -> f16 LDS [BM][LDA_S].
// BM*KC/4 = 1024 quads = 4 per thread; all loads issued, then convert+store.
__device__ __forceinline__ void stage_a_f32(const float* __restrict__ X, size_t lda,
                                            int rowBase, int kb, _Float16* As) {
  v4f s[4];
#pragma unroll
  for (int it = 0; it < 4; ++it) {
    int i = threadIdx.x + it * NTHREADS;
    int rr = i >> 3;            // KC/4 = 8 quads per row
    int k4 = (i & 7) * 4;
    s[it] = *(const v4f*)&X[(size_t)(rowBase + rr) * lda + kb + k4];
  }
#pragma unroll
  for (int it = 0; it < 4; ++it) {
    int i = threadIdx.x + it * NTHREADS;
    int rr = i >> 3;
    int k4 = (i & 7) * 4;
    v4h d;
#pragma unroll
    for (int j = 0; j < 4; ++j) d[j] = (_Float16)s[it][j];
    *(v4h*)&As[(size_t)rr * LDA_S + k4] = d;
  }
}

// Stage B tile (f32, row-major K x N, stride ldw) transposed -> [BN][LDB_T].
// (KC/4)*BN = 1024 quads = 4 per thread; coalesced reads along N.
__device__ __forceinline__ void stage_b_f32(const float* __restrict__ W, size_t ldw,
                                            int kb, int colBase, _Float16* Bs_t) {
  float s[4][4];
#pragma unroll
  for (int it = 0; it < 4; ++it) {
    int i = threadIdx.x + it * NTHREADS;
    int nn = i & (BN - 1);
    int k4 = (i >> 7) * 4;      // i / BN * 4
    const float* p = &W[(size_t)(kb + k4) * ldw + colBase + nn];
#pragma unroll
    for (int j = 0; j < 4; ++j) s[it][j] = p[(size_t)j * ldw];
  }
#pragma unroll
  for (int it = 0; it < 4; ++it) {
    int i = threadIdx.x + it * NTHREADS;
    int nn = i & (BN - 1);
    int k4 = (i >> 7) * 4;
    v4h d;
#pragma unroll
    for (int j = 0; j < 4; ++j) d[j] = (_Float16)s[it][j];
    *(v4h*)&Bs_t[(size_t)nn * LDB_T + k4] = d;
  }
}

// Epilogue coordinates per lane: wm=(wave>>1)*32, wn=(wave&1)*64,
// row = rowBase+wm+tm*16+hlf*8+v ; col = colBase+wn+tn*16+n

// ---------------------------------------------------------------------------
// Kernel 1: h = x @ W_in ; split into xs (f16, conv input) and silu(res) (f32)
// M=2048, N=4096, K=1024
// ---------------------------------------------------------------------------
__global__ __launch_bounds__(NTHREADS)
void k_gemm_in(const float* __restrict__ x, const float* __restrict__ W_in,
               _Float16* __restrict__ xs16, float* __restrict__ sres32) {
  __shared__ _Float16 As[BM * LDA_S];
  __shared__ _Float16 Bs_t[BN * LDB_T];
  const int tid = threadIdx.x;
  const int rowBase = blockIdx.y * BM;
  const int colBase = blockIdx.x * BN;
  v8f acc[2][4] = {};
  for (int kb = 0; kb < F_DIM; kb += KC) {
    stage_a_f32(x, F_DIM, rowBase, kb, As);
    stage_b_f32(W_in, 2 * H_DIM, kb, colBase, Bs_t);
    if (tid < KC && kb + KC < F_DIM)  // hint next B tile -> global_prefetch
      __builtin_prefetch(&W_in[(size_t)(kb + KC + tid) * (2 * H_DIM) + colBase], 0, 1);
    __syncthreads();
    wmma_step(As, Bs_t, acc);
    __syncthreads();
  }
  const int wave = tid >> 5, lane = tid & 31;
  const int wm = (wave >> 1) * 32, wn = (wave & 1) * 64;
  const int hlf = lane >> 4, n = lane & 15;
#pragma unroll
  for (int tm = 0; tm < 2; ++tm)
#pragma unroll
    for (int tn = 0; tn < 4; ++tn)
#pragma unroll
      for (int v = 0; v < 8; ++v) {
        int row = rowBase + wm + tm * 16 + hlf * 8 + v;
        int col = colBase + wn + tn * 16 + n;
        float val = acc[tm][tn][v];
        if (col < H_DIM)
          xs16[(size_t)row * H_DIM + col] = (_Float16)val;
        else
          sres32[(size_t)row * H_DIM + (col - H_DIM)] = siluf(val);
      }
}

// ---------------------------------------------------------------------------
// Kernel 2: grouped causal conv (K=4, groups=2) as 4 accumulated shifted
// GEMMs + bias + SiLU -> u.  M=2048, N=2048, reduction 4*1024.
// ---------------------------------------------------------------------------
__global__ __launch_bounds__(NTHREADS)
void k_conv(const _Float16* __restrict__ xs16, const float* __restrict__ W_conv,
            const float* __restrict__ b_conv, float* __restrict__ u32) {
  __shared__ _Float16 As[BM * LDA_S];
  __shared__ _Float16 Bs_t[BN * LDB_T];
  const int tid = threadIdx.x;
  const int rowBase = blockIdx.y * BM;
  const int colBase = blockIdx.x * BN;
  const int grp = colBase / (H_DIM / 2);     // group of this output tile
  const int aCol = grp * (H_DIM / 2);        // input-channel base of group
  v8f acc[2][4] = {};
  for (int tap = 0; tap < K_CONV; ++tap) {
    const int rshift = (tap - (K_CONV - 1)) * B_DIM;  // time shift in rows
    for (int kb = 0; kb < H_DIM / 2; kb += KC) {
      // A tile: f16 source, b128 copies, zero-fill rows before t=0.
      // BM*KC/8 = 512 octets = 2 per thread.
      v8h av[2];
#pragma unroll
      for (int it = 0; it < 2; ++it) {
        int i = tid + it * NTHREADS;
        int rr = i >> 2;        // KC/8 = 4 octets per row
        int k8 = (i & 3) * 8;
        int rs = rowBase + rr + rshift;
        v8h d = {};
        if (rs >= 0) d = *(const v8h*)&xs16[(size_t)rs * H_DIM + aCol + kb + k8];
        av[it] = d;
      }
#pragma unroll
      for (int it = 0; it < 2; ++it) {
        int i = tid + it * NTHREADS;
        int rr = i >> 2;
        int k8 = (i & 3) * 8;
        *(v8h*)&As[(size_t)rr * LDA_S + k8] = av[it];
      }
      stage_b_f32(W_conv + (size_t)tap * (H_DIM / 2) * H_DIM, H_DIM, kb, colBase, Bs_t);
      if (tid < KC && kb + KC < H_DIM / 2)
        __builtin_prefetch(&W_conv[((size_t)tap * (H_DIM / 2) + kb + KC + tid) * H_DIM + colBase], 0, 1);
      __syncthreads();
      wmma_step(As, Bs_t, acc);
      __syncthreads();
    }
  }
  const int wave = tid >> 5, lane = tid & 31;
  const int wm = (wave >> 1) * 32, wn = (wave & 1) * 64;
  const int hlf = lane >> 4, n = lane & 15;
#pragma unroll
  for (int tm = 0; tm < 2; ++tm)
#pragma unroll
    for (int tn = 0; tn < 4; ++tn)
#pragma unroll
      for (int v = 0; v < 8; ++v) {
        int row = rowBase + wm + tm * 16 + hlf * 8 + v;
        int col = colBase + wn + tn * 16 + n;
        float val = acc[tm][tn][v] + b_conv[col];
        u32[(size_t)row * H_DIM + col] = siluf(val);
      }
}

// ---------------------------------------------------------------------------
// Kernel 3: proj = u @ W_ssm.  M=2048, N=96 (bounds-checked), K=2048
// ---------------------------------------------------------------------------
__global__ __launch_bounds__(NTHREADS)
void k_proj(const float* __restrict__ u32, const float* __restrict__ W_ssm,
            float* __restrict__ proj32) {
  __shared__ _Float16 As[BM * LDA_S];
  __shared__ _Float16 Bs_t[BN * LDB_T];
  const int tid = threadIdx.x;
  const int rowBase = blockIdx.y * BM;
  v8f acc[2][4] = {};
  for (int kb = 0; kb < H_DIM; kb += KC) {
    stage_a_f32(u32, H_DIM, rowBase, kb, As);
    // B tile with N bound (cols >= 96 zero-filled)
    {
      float s[4][4];
#pragma unroll
      for (int it = 0; it < 4; ++it) {
        int i = tid + it * NTHREADS;
        int nn = i & (BN - 1);
        int k4 = (i >> 7) * 4;
#pragma unroll
        for (int j = 0; j < 4; ++j)
          s[it][j] = (nn < PROJ_W) ? W_ssm[(size_t)(kb + k4 + j) * PROJ_W + nn] : 0.0f;
      }
#pragma unroll
      for (int it = 0; it < 4; ++it) {
        int i = tid + it * NTHREADS;
        int nn = i & (BN - 1);
        int k4 = (i >> 7) * 4;
        v4h d;
#pragma unroll
        for (int j = 0; j < 4; ++j) d[j] = (_Float16)s[it][j];
        *(v4h*)&Bs_t[(size_t)nn * LDB_T + k4] = d;
      }
    }
    __syncthreads();
    wmma_step(As, Bs_t, acc);
    __syncthreads();
  }
  const int wave = tid >> 5, lane = tid & 31;
  const int wm = (wave >> 1) * 32, wn = (wave & 1) * 64;
  const int hlf = lane >> 4, n = lane & 15;
#pragma unroll
  for (int tm = 0; tm < 2; ++tm)
#pragma unroll
    for (int tn = 0; tn < 4; ++tn)
#pragma unroll
      for (int v = 0; v < 8; ++v) {
        int row = rowBase + wm + tm * 16 + hlf * 8 + v;
        int col = wn + tn * 16 + n;
        if (col < PROJ_W) proj32[(size_t)row * PROJ_W + col] = acc[tm][tn][v];
      }
}

// ---------------------------------------------------------------------------
// Kernel 4: delta = clamp(softplus(dt @ W_dt + b_dt)).  M=2048, N=2048, K=64
// dt = proj columns 32..95
// ---------------------------------------------------------------------------
__global__ __launch_bounds__(NTHREADS)
void k_dt(const float* __restrict__ proj32, const float* __restrict__ W_dt,
          const float* __restrict__ b_dt, float* __restrict__ delta32) {
  __shared__ _Float16 As[BM * LDA_S];
  __shared__ _Float16 Bs_t[BN * LDB_T];
  const int tid = threadIdx.x;
  const int rowBase = blockIdx.y * BM;
  const int colBase = blockIdx.x * BN;
  v8f acc[2][4] = {};
  for (int kb = 0; kb < R_DIM; kb += KC) {
    stage_a_f32(proj32 + 2 * N_DIM, PROJ_W, rowBase, kb, As);  // dt columns
    stage_b_f32(W_dt, H_DIM, kb, colBase, Bs_t);
    __syncthreads();
    wmma_step(As, Bs_t, acc);
    __syncthreads();
  }
  const int wave = tid >> 5, lane = tid & 31;
  const int wm = (wave >> 1) * 32, wn = (wave & 1) * 64;
  const int hlf = lane >> 4, n = lane & 15;
#pragma unroll
  for (int tm = 0; tm < 2; ++tm)
#pragma unroll
    for (int tn = 0; tn < 4; ++tn)
#pragma unroll
      for (int v = 0; v < 8; ++v) {
        int row = rowBase + wm + tm * 16 + hlf * 8 + v;
        int col = colBase + wn + tn * 16 + n;
        float d = softplusf(acc[tm][tn][v] + b_dt[col]);
        delta32[(size_t)row * H_DIM + col] = fminf(fmaxf(d, 0.001f), 0.1f);
      }
}

// ---------------------------------------------------------------------------
// Chunked selective scan (3 phases). State per (b,h,n); chunk c covers
// l in [c*CLEN, (c+1)*CLEN).  s_{l} = a_l * s_{l-1} + dBu_l.
// ---------------------------------------------------------------------------
__global__ __launch_bounds__(NTHREADS)
void k_scan_chunks(const float* __restrict__ delta32, const float* __restrict__ u32,
                   const float* __restrict__ proj32, const float* __restrict__ A_log,
                   float* __restrict__ aP, float* __restrict__ bA) {
  int t = blockIdx.x * NTHREADS + threadIdx.x;  // layout [c][b][h][n]
  int n = t & (N_DIM - 1);
  int h = (t >> 4) & (H_DIM - 1);
  int b = (t >> 15) & (B_DIM - 1);
  int c = t >> 16;
  const float Ahn = -__expf(A_log[h * N_DIM + n]);
  float a_prod = 1.0f, b_acc = 0.0f;
  const int l0 = c * CLEN;
  for (int s = 0; s < CLEN; ++s) {
    int r = (l0 + s) * B_DIM + b;
    float dlt = delta32[(size_t)r * H_DIM + h];
    float uu  = u32[(size_t)r * H_DIM + h];
    float Bm  = proj32[(size_t)r * PROJ_W + n];
    float a = __expf(dlt * Ahn);
    a_prod *= a;
    b_acc = a * b_acc + dlt * uu * Bm;
  }
  int i = ((b * H_DIM) + h) * N_DIM + n;
  aP[(size_t)c * BHN + i] = a_prod;
  bA[(size_t)c * BHN + i] = b_acc;
}

__global__ __launch_bounds__(NTHREADS)
void k_scan_prefix(const float* __restrict__ aP, const float* __restrict__ bA,
                   float* __restrict__ sInit) {
  int i = blockIdx.x * NTHREADS + threadIdx.x;  // over BHN
  float s = 0.0f;
#pragma unroll
  for (int c = 0; c < NCHUNK; ++c) {
    sInit[(size_t)c * BHN + i] = s;
    s = aP[(size_t)c * BHN + i] * s + bA[(size_t)c * BHN + i];
  }
}

// Phase 3: replay each chunk with correct init; reduce over n with 16-wide
// shfl_xor inside each half of the wave32; y = sum_n Cm*s + u*D
__global__ __launch_bounds__(NTHREADS)
void k_scan_y(const float* __restrict__ delta32, const float* __restrict__ u32,
              const float* __restrict__ proj32, const float* __restrict__ A_log,
              const float* __restrict__ Dv, const float* __restrict__ sInit,
              float* __restrict__ y32) {
  const int bx = blockIdx.x;              // [b][hBlk][c]
  const int c = bx & (NCHUNK - 1);
  const int hBlk = (bx >> 4) & (H_DIM / 16 - 1);
  const int b = bx >> 11;
  const int tid = threadIdx.x;
  const int wave = tid >> 5, lane = tid & 31;
  const int which = lane >> 4, n = lane & 15;
  const int h = hBlk * 16 + wave * 2 + which;
  const float Ahn = -__expf(A_log[h * N_DIM + n]);
  const float Dh = Dv[h];
  const int i = ((b * H_DIM) + h) * N_DIM + n;
  float s = sInit[(size_t)c * BHN + i];
  const int l0 = c * CLEN;
  for (int st = 0; st < CLEN; ++st) {
    int r = (l0 + st) * B_DIM + b;
    float dlt = delta32[(size_t)r * H_DIM + h];
    float uu  = u32[(size_t)r * H_DIM + h];
    float Bm  = proj32[(size_t)r * PROJ_W + n];
    float Cm  = proj32[(size_t)r * PROJ_W + N_DIM + n];
    float a = __expf(dlt * Ahn);
    s = a * s + dlt * uu * Bm;
    float contrib = Cm * s;
    contrib += __shfl_xor(contrib, 1, 16);
    contrib += __shfl_xor(contrib, 2, 16);
    contrib += __shfl_xor(contrib, 4, 16);
    contrib += __shfl_xor(contrib, 8, 16);
    if (n == 0) y32[(size_t)r * H_DIM + h] = contrib + uu * Dh;
  }
}

// ---------------------------------------------------------------------------
// Kernel 8: out = (y * silu(res)) @ W_out.  M=2048, N=1024, K=2048.
// Elementwise product fused into A-tile conversion.
// ---------------------------------------------------------------------------
__global__ __launch_bounds__(NTHREADS)
void k_out(const float* __restrict__ y32, const float* __restrict__ sres32,
           const float* __restrict__ W_out, float* __restrict__ out) {
  __shared__ _Float16 As[BM * LDA_S];
  __shared__ _Float16 Bs_t[BN * LDB_T];
  const int tid = threadIdx.x;
  const int rowBase = blockIdx.y * BM;
  const int colBase = blockIdx.x * BN;
  v8f acc[2][4] = {};
  for (int kb = 0; kb < H_DIM; kb += KC) {
    {
      v4f yv[4], sv[4];
#pragma unroll
      for (int it = 0; it < 4; ++it) {
        int i = tid + it * NTHREADS;
        int rr = i >> 3;
        int k4 = (i & 7) * 4;
        size_t idx = (size_t)(rowBase + rr) * H_DIM + kb + k4;
        yv[it] = *(const v4f*)&y32[idx];
        sv[it] = *(const v4f*)&sres32[idx];
      }
#pragma unroll
      for (int it = 0; it < 4; ++it) {
        int i = tid + it * NTHREADS;
        int rr = i >> 3;
        int k4 = (i & 7) * 4;
        v4h d;
#pragma unroll
        for (int j = 0; j < 4; ++j) d[j] = (_Float16)(yv[it][j] * sv[it][j]);
        *(v4h*)&As[(size_t)rr * LDA_S + k4] = d;
      }
    }
    stage_b_f32(W_out, F_DIM, kb, colBase, Bs_t);
    if (tid < KC && kb + KC < H_DIM)
      __builtin_prefetch(&W_out[(size_t)(kb + KC + tid) * F_DIM + colBase], 0, 1);
    __syncthreads();
    wmma_step(As, Bs_t, acc);
    __syncthreads();
  }
  const int wave = tid >> 5, lane = tid & 31;
  const int wm = (wave >> 1) * 32, wn = (wave & 1) * 64;
  const int hlf = lane >> 4, n = lane & 15;
#pragma unroll
  for (int tm = 0; tm < 2; ++tm)
#pragma unroll
    for (int tn = 0; tn < 4; ++tn)
#pragma unroll
      for (int v = 0; v < 8; ++v) {
        int row = rowBase + wm + tm * 16 + hlf * 8 + v;
        int col = colBase + wn + tn * 16 + n;
        out[(size_t)row * F_DIM + col] = acc[tm][tn][v];
      }
}

// ---------------------------------------------------------------------------
extern "C" void kernel_launch(void* const* d_in, const int* in_sizes, int n_in,
                              void* d_out, int out_size, void* d_ws, size_t ws_size,
                              hipStream_t stream) {
  (void)in_sizes; (void)n_in; (void)out_size; (void)ws_size;
  const float* x      = (const float*)d_in[0];
  const float* W_in   = (const float*)d_in[1];
  const float* W_conv = (const float*)d_in[2];
  const float* b_conv = (const float*)d_in[3];
  const float* A_log  = (const float*)d_in[4];
  const float* Dv     = (const float*)d_in[5];
  const float* W_ssm  = (const float*)d_in[6];
  const float* W_dt   = (const float*)d_in[7];
  const float* b_dt   = (const float*)d_in[8];
  const float* W_out  = (const float*)d_in[9];
  float* out = (float*)d_out;

  char* ws = (char*)d_ws;
  size_t off = 0;
  auto wsalloc = [&](size_t bytes) -> char* {
    size_t start = (off + 255) & ~(size_t)255;
    off = start + bytes;
    return ws + start;
  };
  _Float16* xs16 = (_Float16*)wsalloc((size_t)MROWS * H_DIM * sizeof(_Float16));
  float* sres32  = (float*)wsalloc((size_t)MROWS * H_DIM * sizeof(float));
  float* u32     = (float*)wsalloc((size_t)MROWS * H_DIM * sizeof(float));
  float* proj32  = (float*)wsalloc((size_t)MROWS * PROJ_W * sizeof(float));
  float* delta32 = (float*)wsalloc((size_t)MROWS * H_DIM * sizeof(float));
  float* y32     = (float*)wsalloc((size_t)MROWS * H_DIM * sizeof(float));
  float* aP      = (float*)wsalloc((size_t)NCHUNK * BHN * sizeof(float));
  float* bA      = (float*)wsalloc((size_t)NCHUNK * BHN * sizeof(float));
  float* sInit   = (float*)wsalloc((size_t)NCHUNK * BHN * sizeof(float));

  dim3 blk(NTHREADS);
  k_gemm_in<<<dim3(2 * H_DIM / BN, MROWS / BM), blk, 0, stream>>>(x, W_in, xs16, sres32);
  k_conv<<<dim3(H_DIM / BN, MROWS / BM), blk, 0, stream>>>(xs16, W_conv, b_conv, u32);
  k_proj<<<dim3(1, MROWS / BM), blk, 0, stream>>>(u32, W_ssm, proj32);
  k_dt<<<dim3(H_DIM / BN, MROWS / BM), blk, 0, stream>>>(proj32, W_dt, b_dt, delta32);
  k_scan_chunks<<<dim3(NCHUNK * BHN / NTHREADS), blk, 0, stream>>>(delta32, u32, proj32, A_log, aP, bA);
  k_scan_prefix<<<dim3(BHN / NTHREADS), blk, 0, stream>>>(aP, bA, sInit);
  k_scan_y<<<dim3(B_DIM * (H_DIM / 16) * NCHUNK), blk, 0, stream>>>(delta32, u32, proj32, A_log, Dv, sInit, y32);
  k_out<<<dim3(F_DIM / BN, MROWS / BM), blk, 0, stream>>>(y32, sres32, W_out, out);
}
